// Experts_28123445854551
// MI455X (gfx1250) — matrix operand order repaired and checked by reading, not compiled, not run on hardware
//
#include <hip/hip_runtime.h>

// Problem constants (from reference)
#define N_NODES 20000
#define N_EDGES 320000
#define HDIM    128
#define KEXP    4
#define NLAYER  3
#define NBATCH  64
#define NCLS    2
#define LDB     136   // padded LDS row (bf16 elems) to break bank conflicts
#define HH      (HDIM * HDIM)

typedef __bf16 bf16;
typedef __attribute__((ext_vector_type(16))) __bf16 v16bf;
typedef __attribute__((ext_vector_type(8)))  float  v8f;

static __device__ inline v8f wmma_bf16(v16bf a, v16bf b, v8f c) {
  // D = A(16x32 bf16) x B(32x16 bf16) + C(16x16 f32)
  return __builtin_amdgcn_wmma_f32_16x16x32_bf16(false, a, false, b, (short)0, c,
                                                 false, false);
}

// ---- A fragment (16x32 bf16) built from global f32 row, fused z = s*h + agg ----
// ISA layout: lanes 0-15 row M=lane, K={0..7,16..23}; lanes 16-31 K={8..15,24..31}
static __device__ inline v16bf a_frag_zmix(const float* __restrict__ h,
                                           const float* __restrict__ agg,
                                           float scale, int grow, int k_base,
                                           int valid) {
  const int lane = threadIdx.x & 31;
  const int half = lane >> 4;
  const float m = valid ? 1.0f : 0.0f;
  const size_t base = (size_t)grow * HDIM + k_base + half * 8;
  const float* hp = h + base;
  const float* ap = agg + base;
  v16bf a;
#pragma unroll
  for (int i = 0; i < 8; ++i) {
    const int off = (i < 4) ? (2 * i) : (16 + 2 * (i - 4));
    a[2 * i]     = (bf16)(m * (scale * hp[off]     + ap[off]));
    a[2 * i + 1] = (bf16)(m * (scale * hp[off + 1] + ap[off + 1]));
  }
  return a;
}

// A fragment from a per-lane global f32 row pointer (gathered node features)
static __device__ inline v16bf a_frag_row(const float* __restrict__ rowp,
                                          int k_base, int valid) {
  const int lane = threadIdx.x & 31;
  const int half = lane >> 4;
  const float m = valid ? 1.0f : 0.0f;
  const float* p = rowp + k_base + half * 8;
  v16bf a;
#pragma unroll
  for (int i = 0; i < 8; ++i) {
    const int off = (i < 4) ? (2 * i) : (16 + 2 * (i - 4));
    a[2 * i]     = (bf16)(m * p[off]);
    a[2 * i + 1] = (bf16)(m * p[off + 1]);
  }
  return a;
}

// A fragment from LDS bf16 tile (row-major, stride LDB)
static __device__ inline v16bf a_frag_lds(const bf16* __restrict__ t,
                                          int row_base, int k_base) {
  const int lane = threadIdx.x & 31;
  const int half = lane >> 4;
  const int r = lane & 15;
  const bf16* p = t + (size_t)(row_base + r) * LDB + k_base + half * 8;
  v16bf a;
#pragma unroll
  for (int i = 0; i < 8; ++i) {
    const int off = (i < 4) ? (2 * i) : (16 + 2 * (i - 4));
    a[2 * i]     = p[off];
    a[2 * i + 1] = p[off + 1];
  }
  return a;
}

// B fragment (32x16 bf16) from TRANSPOSED LDS tile wT[col][k] (stride LDB).
// Lane (half,n): col = col_base+n, K = {half*8 + 0..7, 16 + half*8 + 0..7}.
static __device__ inline v16bf b_frag_ldsT(const bf16* __restrict__ wT,
                                           int k_base, int col_base) {
  const int lane = threadIdx.x & 31;
  const int half = lane >> 4;
  const int n = lane & 15;
  const bf16* p = wT + (size_t)(col_base + n) * LDB + k_base + half * 8;
  v16bf b;
#pragma unroll
  for (int i = 0; i < 16; ++i) b[i] = p[(i < 8) ? i : (8 + i)];
  return b;
}

// ---- gfx1250 async copy: global (pre-transposed bf16 weights) -> LDS ----
// Copies one 128x128 bf16 matrix (col-major rows of 256B) into a padded
// (LDB-stride) LDS tile. Tracked by ASYNCcnt; caller must wait_async + barrier.
static __device__ inline void async_load_w_tile(const bf16* __restrict__ gsrc,
                                                bf16* __restrict__ lbuf) {
  const unsigned lbase = (unsigned)(uintptr_t)lbuf;  // low 32 bits = LDS offset
  for (int i = threadIdx.x; i < 2048; i += blockDim.x) {
    const int c = i >> 4;  // column (0..127)
    const int j = i & 15;  // 16-byte chunk within the 256B column row
    const unsigned goff = (unsigned)((c * HDIM + j * 8) * sizeof(bf16));
    const unsigned loff = lbase + (unsigned)((c * LDB + j * 8) * sizeof(bf16));
    asm volatile("global_load_async_to_lds_b128 %0, %1, %2"
                 :
                 : "v"(loff), "v"(goff), "s"(gsrc)
                 : "memory");
  }
}
static __device__ inline void wait_async_all() {
  asm volatile("s_wait_asynccnt 0" ::: "memory");
}

// ---------------- utility kernels ----------------
__global__ void zero_f32_kernel(float* __restrict__ p, int n) {
  int t = blockIdx.x * blockDim.x + threadIdx.x;
  if (t < n) p[t] = 0.0f;
}
__global__ void zero_u32_kernel(unsigned* __restrict__ p, int n) {
  int t = blockIdx.x * blockDim.x + threadIdx.x;
  if (t < n) p[t] = 0u;
}

// Convert nmat consecutive 128x128 f32 matrices into transposed bf16:
// out[m][c*128 + k] = (bf16) W[m][k*128 + c]
__global__ void prep_wT_kernel(const float* __restrict__ W,
                               bf16* __restrict__ out, int nmat) {
  int t = blockIdx.x * blockDim.x + threadIdx.x;
  if (t >= nmat * HH) return;
  int m = t >> 14;
  int idx = t & (HH - 1);
  int k = idx >> 7;
  int c = idx & 127;
  out[(size_t)m * HH + c * HDIM + k] = (bf16)W[t];
}

// agg[dst[e]] += h[src[e]] * w   (32 threads per edge, float4 per thread)
__global__ void scatter_add_kernel(const float* __restrict__ h,
                                   const int* __restrict__ src,
                                   const int* __restrict__ dst,
                                   const float* __restrict__ ew,
                                   float* __restrict__ agg, int nedges) {
  int t = blockIdx.x * blockDim.x + threadIdx.x;
  int e = t >> 5;
  int c = (t & 31) * 4;
  if (e >= nedges) return;
  float w = ew ? ew[e] : 1.0f;
  if (w == 0.0f) return;
  int s = src[e], d = dst[e];
  const float4 v = *(const float4*)(h + (size_t)s * HDIM + c);
  float* o = agg + (size_t)d * HDIM + c;
  atomicAdd(o + 0, v.x * w);
  atomicAdd(o + 1, v.y * w);
  atomicAdd(o + 2, v.z * w);
  atomicAdd(o + 3, v.w * w);
}

// ---------------- fused GIN node MLP: out = relu(relu(z@W1+b1)@W2+b2) ----------------
// z = (1+eps)*h + agg. 64-row tile per block, 4 waves. Weights are
// pre-transposed bf16 in global; staged via async-to-LDS, double-buffered.
__global__ __launch_bounds__(128)
void gin_mlp_kernel(const float* __restrict__ h, const float* __restrict__ agg,
                    const bf16* __restrict__ wt1, const float* __restrict__ b1,
                    const bf16* __restrict__ wt2, const float* __restrict__ b2,
                    const float* __restrict__ eps_arr, int eps_idx, int nrows,
                    float* __restrict__ out) {
  __shared__ bf16 wbuf[2][HDIM * LDB];  // double-buffered weights [col][k]
  __shared__ bf16 yt[64 * LDB];         // relu(GEMM1) staging
  __shared__ float bias1[HDIM];
  __shared__ float bias2[HDIM];

  const float scale = 1.0f + eps_arr[eps_idx];
  const int row0 = blockIdx.x * 64;
  const int wave = threadIdx.x >> 5;
  const int lane = threadIdx.x & 31;
  const int half = lane >> 4;
  const int n = lane & 15;
  const int wrow = wave * 16;

  async_load_w_tile(wt1, wbuf[0]);
  if (threadIdx.x < HDIM) {
    bias1[threadIdx.x] = b1[threadIdx.x];
    bias2[threadIdx.x] = b2[threadIdx.x];
  }
  wait_async_all();
  __syncthreads();

  // Prefetch W2 into the second buffer while GEMM1 runs (ASYNCcnt in flight).
  async_load_w_tile(wt2, wbuf[1]);

  v8f acc[8];
#pragma unroll
  for (int nt = 0; nt < 8; ++nt) acc[nt] = (v8f){0, 0, 0, 0, 0, 0, 0, 0};

  // GEMM1: A fused from global, B from LDS
  {
    const int r = lane & 15;
    const int grow = row0 + wrow + r;
    const int valid = grow < nrows;
    const int grc = valid ? grow : 0;
#pragma unroll
    for (int kt = 0; kt < 4; ++kt) {
      v16bf a = a_frag_zmix(h, agg, scale, grc, kt * 32, valid);
#pragma unroll
      for (int nt = 0; nt < 8; ++nt)
        acc[nt] = wmma_bf16(a, b_frag_ldsT(wbuf[0], kt * 32, nt * 16), acc[nt]);
    }
  }

  // bias + relu -> yt (bf16), C/D layout: VGPR i = row half*8+i, col = lane n
#pragma unroll
  for (int nt = 0; nt < 8; ++nt) {
    const int col = nt * 16 + n;
    const float bv = bias1[col];
#pragma unroll
    for (int i = 0; i < 8; ++i) {
      float v = acc[nt][i] + bv;
      yt[(wrow + half * 8 + i) * LDB + col] = (bf16)(v > 0.0f ? v : 0.0f);
    }
  }
  wait_async_all();  // W2 copy complete (this wave's issues)
  __syncthreads();   // all waves' copies + yt visible

#pragma unroll
  for (int nt = 0; nt < 8; ++nt) acc[nt] = (v8f){0, 0, 0, 0, 0, 0, 0, 0};
#pragma unroll
  for (int kt = 0; kt < 4; ++kt) {
    v16bf a = a_frag_lds(yt, wrow, kt * 32);
#pragma unroll
    for (int nt = 0; nt < 8; ++nt)
      acc[nt] = wmma_bf16(a, b_frag_ldsT(wbuf[1], kt * 32, nt * 16), acc[nt]);
  }

#pragma unroll
  for (int nt = 0; nt < 8; ++nt) {
    const int col = nt * 16 + n;
    const float bv = bias2[col];
#pragma unroll
    for (int i = 0; i < 8; ++i) {
      const int grow = row0 + wrow + half * 8 + i;
      if (grow < nrows) {
        float v = acc[nt][i] + bv;
        out[(size_t)grow * HDIM + col] = v > 0.0f ? v : 0.0f;
      }
    }
  }
}

// ---------------- edge mask MLP + gumbel hard threshold ----------------
// logit = relu([Z[src],Z[dst]] @ W1 + b1) . w2 + b2 ; e_on = (logit + g > 0)
// W1 K-halves pre-transposed bf16 in global; double-buffered async staging.
__global__ __launch_bounds__(128)
void mask_mlp_kernel(const float* __restrict__ Z, const int* __restrict__ src,
                     const int* __restrict__ dst, const float* __restrict__ u,
                     const bf16* __restrict__ w1h0, const bf16* __restrict__ w1h1,
                     const float* __restrict__ mb1, const float* __restrict__ mw2,
                     const float* __restrict__ mb2, int kexp,
                     float* __restrict__ e_on, float* __restrict__ emask_out,
                     unsigned* __restrict__ nw, int nedges) {
  __shared__ bf16 wbuf[2][HDIM * LDB];
  __shared__ float bias1[HDIM];
  __shared__ float w2v[HDIM];

  const int e0 = blockIdx.x * 64;
  const int wave = threadIdx.x >> 5;
  const int lane = threadIdx.x & 31;
  const int half = lane >> 4;
  const int n = lane & 15;
  const int wrow = wave * 16;

  async_load_w_tile(w1h0, wbuf[0]);
  if (threadIdx.x < HDIM) {
    bias1[threadIdx.x] = mb1[threadIdx.x];
    w2v[threadIdx.x] = mw2[threadIdx.x];
  }
  wait_async_all();
  __syncthreads();
  async_load_w_tile(w1h1, wbuf[1]);  // overlap with first K-half GEMM

  v8f acc[8];
#pragma unroll
  for (int nt = 0; nt < 8; ++nt) acc[nt] = (v8f){0, 0, 0, 0, 0, 0, 0, 0};

  const int r = lane & 15;
  const int ge = e0 + wrow + r;
  const int valid = ge < nedges;

  // K-half 0: rows gathered from Z[src]
  {
    const int node = valid ? src[ge] : 0;
    const float* zrow = Z + (size_t)node * HDIM;
#pragma unroll
    for (int kt = 0; kt < 4; ++kt) {
      v16bf a = a_frag_row(zrow, kt * 32, valid);
#pragma unroll
      for (int nt = 0; nt < 8; ++nt)
        acc[nt] = wmma_bf16(a, b_frag_ldsT(wbuf[0], kt * 32, nt * 16), acc[nt]);
    }
  }
  wait_async_all();
  __syncthreads();
  // K-half 1: rows gathered from Z[dst]
  {
    const int node = valid ? dst[ge] : 0;
    const float* zrow = Z + (size_t)node * HDIM;
#pragma unroll
    for (int kt = 0; kt < 4; ++kt) {
      v16bf a = a_frag_row(zrow, kt * 32, valid);
#pragma unroll
      for (int nt = 0; nt < 8; ++nt)
        acc[nt] = wmma_bf16(a, b_frag_ldsT(wbuf[1], kt * 32, nt * 16), acc[nt]);
    }
  }

  // layer2: per-lane partial dot with w2, reduce across the 16 lanes of the half
  float part[8];
#pragma unroll
  for (int i = 0; i < 8; ++i) part[i] = 0.0f;
#pragma unroll
  for (int nt = 0; nt < 8; ++nt) {
    const int col = nt * 16 + n;
    const float w2c = w2v[col];
    const float bv = bias1[col];
#pragma unroll
    for (int i = 0; i < 8; ++i) {
      float v = acc[nt][i] + bv;
      part[i] += (v > 0.0f ? v : 0.0f) * w2c;
    }
  }
#pragma unroll
  for (int s = 1; s < 16; s <<= 1) {
#pragma unroll
    for (int i = 0; i < 8; ++i) part[i] += __shfl_xor(part[i], s, 32);
  }

  if (n == 0) {
    const float b2v = mb2[0];
    for (int i = 0; i < 8; ++i) {
      const int e = e0 + wrow + half * 8 + i;
      if (e < nedges) {
        const float logit = part[i] + b2v;
        const float uu = u[(size_t)e * KEXP + kexp];
        const float g = -logf(-logf(uu + 1e-20f) + 1e-20f);
        const float on = (logit + g > 0.0f) ? 1.0f : 0.0f;
        e_on[e] = on;
        emask_out[(size_t)e * KEXP + kexp] = on;
        if (on > 0.0f) {
          atomicOr(&nw[src[e]], 1u);
          atomicOr(&nw[dst[e]], 1u);
        }
      }
    }
  }
}

// masked_x = x * (nw>0); also write node_masks[:,k]
__global__ void apply_node_mask_kernel(const float* __restrict__ x,
                                       const unsigned* __restrict__ nw,
                                       float* __restrict__ mx,
                                       float* __restrict__ nmask_out, int kexp,
                                       int nnodes) {
  int t = blockIdx.x * blockDim.x + threadIdx.x;
  int node = t >> 5;
  int c = (t & 31) * 4;
  if (node >= nnodes) return;
  float m = nw[node] ? 1.0f : 0.0f;
  float4 v = *(const float4*)(x + (size_t)node * HDIM + c);
  v.x *= m; v.y *= m; v.z *= m; v.w *= m;
  *(float4*)(mx + (size_t)node * HDIM + c) = v;
  if (c == 0) nmask_out[(size_t)node * KEXP + kexp] = m;
}

// segment-sum pool into (B x 128) sums + counts
__global__ void pool_accum_kernel(const float* __restrict__ h,
                                  const int* __restrict__ batch,
                                  float* __restrict__ sums,
                                  float* __restrict__ cnts, int nnodes) {
  int t = blockIdx.x * blockDim.x + threadIdx.x;
  int node = t >> 5;
  int c = (t & 31) * 4;
  if (node >= nnodes) return;
  int b = batch[node];
  const float4 v = *(const float4*)(h + (size_t)node * HDIM + c);
  float* o = sums + (size_t)b * HDIM + c;
  atomicAdd(o + 0, v.x);
  atomicAdd(o + 1, v.y);
  atomicAdd(o + 2, v.z);
  atomicAdd(o + 3, v.w);
  if (c == 0) atomicAdd(&cnts[b], 1.0f);
}

__global__ void pool_finalize_kernel(const float* __restrict__ sums,
                                     const float* __restrict__ cnts,
                                     float* __restrict__ out, int row_stride) {
  int t = blockIdx.x * blockDim.x + threadIdx.x;
  if (t >= NBATCH * HDIM) return;
  int b = t >> 7, j = t & 127;
  float c = cnts[b];
  c = c > 1.0f ? c : 1.0f;
  out[(size_t)b * row_stride + j] = sums[t] / c;
}

// head layer1: y = relu((sums/cnt) @ W1 + b1)   (64x128 @ 128x128 — tiny)
__global__ void head1_kernel(const float* __restrict__ sums,
                             const float* __restrict__ cnts,
                             const float* __restrict__ W1,
                             const float* __restrict__ b1,
                             float* __restrict__ y) {
  int t = blockIdx.x * blockDim.x + threadIdx.x;
  if (t >= NBATCH * HDIM) return;
  int b = t >> 7, j = t & 127;
  float c = cnts[b];
  c = c > 1.0f ? c : 1.0f;
  const float inv = 1.0f / c;
  float s = b1[j];
  for (int i = 0; i < HDIM; ++i) s += (sums[b * HDIM + i] * inv) * W1[i * HDIM + j];
  y[t] = s > 0.0f ? s : 0.0f;
}

// head layer2: logits[b, kexp, c] = y @ W2 + b2
__global__ void head2_kernel(const float* __restrict__ y,
                             const float* __restrict__ W2,
                             const float* __restrict__ b2,
                             float* __restrict__ logits, int kexp) {
  int t = blockIdx.x * blockDim.x + threadIdx.x;
  if (t >= NBATCH * NCLS) return;
  int b = t >> 1, c = t & 1;
  float s = b2[c];
  for (int i = 0; i < HDIM; ++i) s += y[b * HDIM + i] * W2[i * NCLS + c];
  logits[((size_t)b * KEXP + kexp) * NCLS + c] = s;
}

// ---------------- orchestration ----------------
extern "C" void kernel_launch(void* const* d_in, const int* in_sizes, int n_in,
                              void* d_out, int out_size, void* d_ws,
                              size_t ws_size, hipStream_t stream) {
  const float* x       = (const float*)d_in[0];
  const int*   ei      = (const int*)d_in[1];
  const int*   batch   = (const int*)d_in[2];
  const float* u       = (const float*)d_in[3];
  const float* enc_W1  = (const float*)d_in[4];
  const float* enc_b1  = (const float*)d_in[5];
  const float* enc_W2  = (const float*)d_in[6];
  const float* enc_b2  = (const float*)d_in[7];
  const float* enc_eps = (const float*)d_in[8];
  const float* cls_W1  = (const float*)d_in[9];
  const float* cls_b1  = (const float*)d_in[10];
  const float* cls_W2  = (const float*)d_in[11];
  const float* cls_b2  = (const float*)d_in[12];
  const float* cls_eps = (const float*)d_in[13];
  const float* mask_W1 = (const float*)d_in[14];
  const float* mask_b1 = (const float*)d_in[15];
  const float* mask_W2 = (const float*)d_in[16];
  const float* mask_b2 = (const float*)d_in[17];
  const float* head_W1 = (const float*)d_in[18];
  const float* head_b1 = (const float*)d_in[19];
  const float* head_W2 = (const float*)d_in[20];
  const float* head_b2 = (const float*)d_in[21];

  const int* src = ei;
  const int* dstp = ei + N_EDGES;

  const int NF = N_NODES * HDIM;  // 2,560,000

  float* ws = (float*)d_ws;
  float* hA   = ws;                    // N x H
  float* hB   = hA + NF;               // N x H
  float* agg  = hB + NF;               // N x H
  float* e_on = agg + NF;              // E
  float* m0   = e_on + N_EDGES;        // N x H
  float* m1   = m0 + NF;               // N x H
  float* psum = m1 + NF;               // B x H
  float* pcnt = psum + NBATCH * HDIM;  // B
  float* yhd  = pcnt + NBATCH;         // B x H
  unsigned* nw = (unsigned*)(yhd + NBATCH * HDIM);  // N
  bf16* wbf = (bf16*)(nw + N_NODES);   // 38 * 16384 bf16 transposed weights

  float* out = (float*)d_out;
  float* out_logits = out;                                // (B,K,C) 512
  float* out_hstab  = out + NBATCH * KEXP * NCLS;         // (B,K,H) 32768
  float* out_horig  = out_hstab + NBATCH * KEXP * HDIM;   // (B,H)   8192
  float* out_nmask  = out_horig + NBATCH * HDIM;          // (N,K,1) 80000
  float* out_emask  = out_nmask + N_NODES * KEXP;         // (E,K,1) 1280000

  const int zb_nf   = (NF + 255) / 256;
  const int sc_blk  = (N_EDGES * 32 + 255) / 256;
  const int mlp_blk = (N_NODES + 63) / 64;
  const int msk_blk = (N_EDGES + 63) / 64;
  const int nm_blk  = (N_NODES * 32 + 255) / 256;
  const int pf_blk  = (NBATCH * HDIM + 255) / 256;

  // -------- one-shot: pre-transpose + bf16-convert all GEMM weights --------
  // layout in wbf (each 16384 elems): [0,3)=enc_W1, [3,6)=enc_W2,
  // [6,18)=cls_W1, [18,30)=cls_W2, [30,38)=mask_W1 (8 K-halves)
  prep_wT_kernel<<<(3 * HH + 255) / 256, 256, 0, stream>>>(enc_W1, wbf, 3);
  prep_wT_kernel<<<(3 * HH + 255) / 256, 256, 0, stream>>>(enc_W2, wbf + 3 * HH, 3);
  prep_wT_kernel<<<(12 * HH + 255) / 256, 256, 0, stream>>>(cls_W1, wbf + 6 * HH, 12);
  prep_wT_kernel<<<(12 * HH + 255) / 256, 256, 0, stream>>>(cls_W2, wbf + 18 * HH, 12);
  prep_wT_kernel<<<(8 * HH + 255) / 256, 256, 0, stream>>>(mask_W1, wbf + 30 * HH, 8);

  // -------- encoder GIN stack (ew == 1) --------
  const float* hin = x;
  float* hout = hA;
  for (int l = 0; l < NLAYER; ++l) {
    zero_f32_kernel<<<zb_nf, 256, 0, stream>>>(agg, NF);
    scatter_add_kernel<<<sc_blk, 256, 0, stream>>>(hin, src, dstp, nullptr, agg,
                                                   N_EDGES);
    gin_mlp_kernel<<<mlp_blk, 128, 0, stream>>>(
        hin, agg, wbf + l * HH, enc_b1 + l * HDIM, wbf + (3 + l) * HH,
        enc_b2 + l * HDIM, enc_eps, l, N_NODES, hout);
    hin = hout;
    hout = (hout == hA) ? hB : hA;
  }
  const float* Z = hin;  // == hA after 3 layers

  // h_orig = mean pool of Z
  zero_f32_kernel<<<pf_blk, 256, 0, stream>>>(psum, NBATCH * HDIM);
  zero_f32_kernel<<<1, 256, 0, stream>>>(pcnt, NBATCH);
  pool_accum_kernel<<<nm_blk, 256, 0, stream>>>(Z, batch, psum, pcnt, N_NODES);
  pool_finalize_kernel<<<pf_blk, 256, 0, stream>>>(psum, pcnt, out_horig, HDIM);

  // -------- experts --------
  for (int k = 0; k < KEXP; ++k) {
    zero_u32_kernel<<<(N_NODES + 255) / 256, 256, 0, stream>>>(nw, N_NODES);
    mask_mlp_kernel<<<msk_blk, 128, 0, stream>>>(
        Z, src, dstp, u, wbf + (30 + 2 * k) * HH, wbf + (31 + 2 * k) * HH,
        mask_b1 + k * HDIM, mask_W2 + k * HDIM, mask_b2 + k, k, e_on, out_emask,
        nw, N_EDGES);
    apply_node_mask_kernel<<<nm_blk, 256, 0, stream>>>(x, nw, m0, out_nmask, k,
                                                       N_NODES);
    const float* hi = m0;
    float* ho = m1;
    for (int l = 0; l < NLAYER; ++l) {
      zero_f32_kernel<<<zb_nf, 256, 0, stream>>>(agg, NF);
      scatter_add_kernel<<<sc_blk, 256, 0, stream>>>(hi, src, dstp, e_on, agg,
                                                     N_EDGES);
      const int kl = k * NLAYER + l;
      gin_mlp_kernel<<<mlp_blk, 128, 0, stream>>>(
          hi, agg, wbf + (6 + kl) * HH, cls_b1 + kl * HDIM, wbf + (18 + kl) * HH,
          cls_b2 + kl * HDIM, cls_eps, kl, N_NODES, ho);
      hi = ho;
      ho = (ho == m1) ? m0 : m1;
    }
    // hi == m1 after 3 layers
    zero_f32_kernel<<<pf_blk, 256, 0, stream>>>(psum, NBATCH * HDIM);
    zero_f32_kernel<<<1, 256, 0, stream>>>(pcnt, NBATCH);
    pool_accum_kernel<<<nm_blk, 256, 0, stream>>>(hi, batch, psum, pcnt,
                                                  N_NODES);
    pool_finalize_kernel<<<pf_blk, 256, 0, stream>>>(psum, pcnt,
                                                     out_hstab + k * HDIM,
                                                     KEXP * HDIM);
    head1_kernel<<<pf_blk, 256, 0, stream>>>(psum, pcnt, head_W1 + k * HH,
                                             head_b1 + k * HDIM, yhd);
    head2_kernel<<<1, 128, 0, stream>>>(yhd, head_W2 + k * HDIM * NCLS,
                                        head_b2 + k * NCLS, out_logits, k);
  }
}
